// FeedForward_15333033246886
// MI455X (gfx1250) — matrix operand order repaired and checked by reading, not compiled, not run on hardware
//
#include <hip/hip_runtime.h>
#include <hip/hip_bf16.h>

// MoE FFN: B=2,S=1024 -> T=2048 tokens, D=512, H=2048, E=8, K=2 (P=4096 pairs)
// Sparse grouped-GEMM in f16 WMMA with f32 accumulation.
// CDNA5 paths: v_wmma_f32_16x16x32_f16, global_load_async_to_lds_b128,
// s_wait_asynccnt, global_prefetch_b8, native v_tanh_f32 for gelu.

typedef _Float16 v2h  __attribute__((ext_vector_type(2)));
typedef _Float16 v4h  __attribute__((ext_vector_type(4)));
typedef _Float16 v8h  __attribute__((ext_vector_type(8)));
typedef _Float16 v16h __attribute__((ext_vector_type(16)));
typedef float    v8f  __attribute__((ext_vector_type(8)));
typedef int      gvec4i __attribute__((vector_size(16)));   // matches builtin param type

#define NT 2048          // tokens
#define NP 4096          // (token, k) pairs
#define ND 512           // model dim
#define NH 2048          // hidden dim
#define NE 8             // experts

#define BM 128
#define BN 128
#define BK 32
#define LDA 40           // padded LDS row stride (f16) for A tile
#define LDB 40           // padded LDS row stride (f16) for B^T tile

#define CAT16(lo, hi) __builtin_shufflevector((lo), (hi), 0,1,2,3,4,5,6,7,8,9,10,11,12,13,14,15)

#if __has_builtin(__builtin_amdgcn_global_load_async_to_lds_b128)
#define HAVE_ASYNC 1
#else
#define HAVE_ASYNC 0
#endif

// generic -> AS1/AS3 via integer round-trip (LDS generic addr low 32 bits = DS offset)
#define AS1P(p) ((__attribute__((address_space(1))) gvec4i*)(unsigned long long)(p))
#define AS3P(p) ((__attribute__((address_space(3))) gvec4i*)(unsigned int)(unsigned long long)(p))

__device__ __forceinline__ void wait_async_all() {
#if HAVE_ASYNC
#if __has_builtin(__builtin_amdgcn_s_wait_asynccnt)
  __builtin_amdgcn_s_wait_asynccnt(0);
#else
  asm volatile("s_wait_asynccnt 0" ::: "memory");
#endif
#endif
}

__device__ __forceinline__ float fast_tanh(float u) {
#if __has_builtin(__builtin_amdgcn_tanhf)
  return __builtin_amdgcn_tanhf(u);
#elif __has_builtin(__builtin_amdgcn_tanh_f32)
  return __builtin_amdgcn_tanh_f32(u);
#else
  // branch-free: tanh(u) = sign(u) * (1 - 2e/(1+e)), e = exp(-2|u|)
  const float a = __builtin_fabsf(u);
  const float em = __expf(-2.0f * a);
  const float t = 1.0f - 2.0f * em / (1.0f + em);
  return u >= 0.0f ? t : -t;
#endif
}

__device__ __forceinline__ float gelu_tanh(float x) {
  // jax.nn.gelu default (approximate=True)
  const float u = 0.7978845608028654f * (x + 0.044715f * x * x * x);
  return 0.5f * x * (1.0f + fast_tanh(u));
}

// -------- workspace layout (bytes) --------
// offs[9]        @ 0
// rowtok[4096]   @ 256
// pairslot[4096] @ 16640
// Xg f16 [4096,512]  @ 33024     (4 MiB)
// Hb f16 [4096,2048] @ 4227328   (16 MiB)
// Ob f32 [4096,512]  @ 21004544  (8 MiB)   total ~28 MiB

// ---------------------------------------------------------------------------
// 1) Routing: per-expert histogram, prefix sum, scatter pair->segment slot
// ---------------------------------------------------------------------------
__global__ __launch_bounds__(256) void k_route(const int* __restrict__ idx,
                                               int* __restrict__ offs,
                                               int* __restrict__ rowtok,
                                               int* __restrict__ pairslot) {
  __shared__ int cnt[NE];
  __shared__ int cur[NE];
  const int tid = threadIdx.x;
  if (tid < NE) cnt[tid] = 0;
  __syncthreads();
  for (int p = tid; p < NP; p += 256) atomicAdd(&cnt[idx[p]], 1);
  __syncthreads();
  if (tid == 0) {
    int s = 0;
    for (int e = 0; e < NE; ++e) { cur[e] = s; offs[e] = s; s += cnt[e]; }
    offs[NE] = s;
  }
  __syncthreads();
  for (int p = tid; p < NP; p += 256) {
    const int e = idx[p];
    const int pos = atomicAdd(&cur[e], 1);
    rowtok[pos] = p >> 1;        // token id
    pairslot[p] = pos;           // inverse map for the combine pass
  }
}

// ---------------------------------------------------------------------------
// 2) Gather token rows -> f16, grouped by expert segment
// ---------------------------------------------------------------------------
__global__ __launch_bounds__(256) void k_gather(const float* __restrict__ x,
                                                const int* __restrict__ rowtok,
                                                _Float16* __restrict__ Xg) {
  const int row = blockIdx.x;               // 0..NP-1
  const int tok = rowtok[row];
  const float2* src = (const float2*)(x + (size_t)tok * ND);
  const float2 v = src[threadIdx.x];        // 256 thr * 2 = 512
  v2h h; h.x = (_Float16)v.x; h.y = (_Float16)v.y;
  *(v2h*)(Xg + (size_t)row * ND + 2 * threadIdx.x) = h;
}

// ---------------------------------------------------------------------------
// 3/4) Grouped GEMM core: C = A_seg[e] (f16) x W[e] (f32->f16) (+bias)
//      GELU=true : write f16 (hidden);  GELU=false : write f32 (pair outputs)
//      A: [NP, KDIM] f16, segment rows per expert.
//      Bsrc: [KDIM, NE, NDIM] f32 (keys / values layout).
// Tiling: 128x128x32 block, 8 wave32 waves (2x4), each wave 64x32 = 8 WMMAs.
// LDS double-buffered; A tile staged with async-to-LDS, B converted in-VALU.
// ---------------------------------------------------------------------------
template <int KDIM, int NDIM, bool GELU>
__global__ __launch_bounds__(256) void k_moe_gemm(const _Float16* __restrict__ A,
                                                  const float* __restrict__ Bsrc,
                                                  const float* __restrict__ bias,
                                                  const int* __restrict__ offs,
                                                  void* __restrict__ outp) {
  const int e    = blockIdx.z;
  const int seg0 = offs[e];
  const int cnt  = offs[e + 1] - seg0;
  const int mBase = blockIdx.x * BM;
  if (mBase >= cnt) return;                 // over-launched tail tile
  const int nBase = blockIdx.y * BN;

  __shared__ _Float16 As[2 * BM * LDA];     // [m][k], padded, double buffer
  __shared__ _Float16 Bt[2 * BN * LDB];     // [n][k], transposed+padded, dbl

  const int tid  = threadIdx.x;
  const int lane = tid & 31;
  const int wv   = tid >> 5;                // 8 waves
  const int wm   = wv >> 2;                 // 0..1  (rows)
  const int wn   = wv & 3;                  // 0..3  (cols)
  const int lh   = lane >> 4;               // lane half
  const int l16  = lane & 15;

  v8f acc[4][2];
#pragma unroll
  for (int mf = 0; mf < 4; ++mf)
#pragma unroll
    for (int nf = 0; nf < 2; ++nf) acc[mf][nf] = (v8f){};

  // ---- tile staging: A via async b128 copies, B via f32->f16 transpose ----
  auto stage = [&](int k0, int buf) {
    _Float16* Asb = As + buf * (BM * LDA);
    _Float16* Btb = Bt + buf * (BN * LDB);
    // A tile: 128 rows x 32 k of f16 -> 512 x 16B chunks, 2 per thread
#pragma unroll
    for (int it = 0; it < 2; ++it) {
      const int c  = tid + it * 256;
      const int r  = c >> 2;
      const int cc = c & 3;
      const int gp = mBase + r;
      const int gpos = seg0 + (gp < cnt ? gp : 0);  // clamp; masked at store
      const _Float16* gsrc = A + (size_t)gpos * KDIM + k0 + cc * 8;
      _Float16* ldst = Asb + r * LDA + cc * 8;
#if HAVE_ASYNC
      __builtin_amdgcn_global_load_async_to_lds_b128(AS1P(gsrc), AS3P(ldst), 0, 0);
#else
      *(v8h*)ldst = *(const v8h*)gsrc;
#endif
    }
    // B tile: 32 k x 128 n f32, each thread: 4 rows x 4 cols sub-block,
    // transpose+convert, four 8B LDS stores.
    const int n4 = (tid & 31) << 2;          // n offset within tile
    const int d  = (tid >> 5) << 2;          // k offset within tile
    const float* gb = Bsrc + ((size_t)(k0 + d) * NE + e) * NDIM + nBase + n4;
    const size_t krow = (size_t)NE * NDIM;
    const float4 r0 = *(const float4*)(gb);
    const float4 r1 = *(const float4*)(gb + krow);
    const float4 r2 = *(const float4*)(gb + 2 * krow);
    const float4 r3 = *(const float4*)(gb + 3 * krow);
    if (k0 + 2 * BK < KDIM)                  // deep prefetch, 2 K-steps ahead
      __builtin_prefetch(gb + (size_t)(2 * BK) * krow, 0, 0);
    _Float16* bb = Btb + n4 * LDB + d;
    *(v4h*)(bb + 0 * LDB) = (v4h){(_Float16)r0.x, (_Float16)r1.x, (_Float16)r2.x, (_Float16)r3.x};
    *(v4h*)(bb + 1 * LDB) = (v4h){(_Float16)r0.y, (_Float16)r1.y, (_Float16)r2.y, (_Float16)r3.y};
    *(v4h*)(bb + 2 * LDB) = (v4h){(_Float16)r0.z, (_Float16)r1.z, (_Float16)r2.z, (_Float16)r3.z};
    *(v4h*)(bb + 3 * LDB) = (v4h){(_Float16)r0.w, (_Float16)r1.w, (_Float16)r2.w, (_Float16)r3.w};
  };

  constexpr int NKI = KDIM / BK;
  stage(0, 0);
  wait_async_all();
  __syncthreads();

  for (int kk = 0; kk < NKI; ++kk) {
    const int cur = kk & 1;
    if (kk + 1 < NKI) stage((kk + 1) * BK, cur ^ 1);   // overlap with compute

    const _Float16* Asb = As + cur * (BM * LDA);
    const _Float16* Btb = Bt + cur * (BN * LDB);

    // A fragments: lanes 0-15 rows M; K octets {lh*8, 16+lh*8}
    v16h af[4];
#pragma unroll
    for (int mf = 0; mf < 4; ++mf) {
      const _Float16* ap = Asb + (wm * 64 + mf * 16 + l16) * LDA + lh * 8;
      const v8h lo = *(const v8h*)ap;
      const v8h hi = *(const v8h*)(ap + 16);
      af[mf] = CAT16(lo, hi);
    }
    // B fragments: lanes 0-15 cols N; K octets {lh*8, 16+lh*8}
#pragma unroll
    for (int nf = 0; nf < 2; ++nf) {
      const _Float16* bp = Btb + (wn * 32 + nf * 16 + l16) * LDB + lh * 8;
      const v8h lo = *(const v8h*)bp;
      const v8h hi = *(const v8h*)(bp + 16);
      const v16h bf = CAT16(lo, hi);
#pragma unroll
      for (int mf = 0; mf < 4; ++mf) {
        acc[mf][nf] = __builtin_amdgcn_wmma_f32_16x16x32_f16(
            false, af[mf], false, bf, (short)0, acc[mf][nf], false, false);
      }
    }
    wait_async_all();
    __syncthreads();
  }

  // ---- epilogue: C lane L, vgpr j -> row = j + 8*lh, col = l16 ----
#pragma unroll
  for (int mf = 0; mf < 4; ++mf) {
#pragma unroll
    for (int nf = 0; nf < 2; ++nf) {
      const int n  = nBase + wn * 32 + nf * 16 + l16;
      const float bv = bias[e * NDIM + n];
#pragma unroll
      for (int j = 0; j < 8; ++j) {
        const int rloc = mBase + wm * 64 + mf * 16 + lh * 8 + j;
        if (rloc < cnt) {
          float v = acc[mf][nf][j] + bv;
          if constexpr (GELU) {
            v = gelu_tanh(v);
            ((_Float16*)outp)[(size_t)(seg0 + rloc) * NDIM + n] = (_Float16)v;
          } else {
            ((float*)outp)[(size_t)(seg0 + rloc) * NDIM + n] = v;
          }
        }
      }
    }
  }
}

// ---------------------------------------------------------------------------
// 5) Combine: out[t] = w0 * Ob[slot0] + w1 * Ob[slot1]  (overwrites out)
// ---------------------------------------------------------------------------
__global__ __launch_bounds__(256) void k_combine(const float* __restrict__ Ob,
                                                 const int* __restrict__ pairslot,
                                                 const float* __restrict__ weights,
                                                 float* __restrict__ out) {
  const int t  = blockIdx.x;                // 0..NT-1
  const int s0 = pairslot[2 * t + 0];
  const int s1 = pairslot[2 * t + 1];
  const float w0 = weights[2 * t + 0];
  const float w1 = weights[2 * t + 1];
  const float2* a = (const float2*)(Ob + (size_t)s0 * ND);
  const float2* b = (const float2*)(Ob + (size_t)s1 * ND);
  float2* o = (float2*)(out + (size_t)t * ND);
  const float2 va = a[threadIdx.x];
  const float2 vb = b[threadIdx.x];
  float2 r;
  r.x = w0 * va.x + w1 * vb.x;
  r.y = w0 * va.y + w1 * vb.y;
  o[threadIdx.x] = r;
}

extern "C" void kernel_launch(void* const* d_in, const int* in_sizes, int n_in,
                              void* d_out, int out_size, void* d_ws, size_t ws_size,
                              hipStream_t stream) {
  const float* x          = (const float*)d_in[0];
  const int*   indices    = (const int*)d_in[1];
  const float* weights    = (const float*)d_in[2];
  const float* keys       = (const float*)d_in[3];   // [D, E, H]
  const float* key_bias   = (const float*)d_in[4];   // [E, H]
  const float* values     = (const float*)d_in[5];   // [H, E, D]
  const float* value_bias = (const float*)d_in[6];   // [E, D]
  float* out = (float*)d_out;

  char* w = (char*)d_ws;
  int*       offs     = (int*)(w + 0);
  int*       rowtok   = (int*)(w + 256);
  int*       pairslot = (int*)(w + 16640);
  _Float16*  Xg       = (_Float16*)(w + 33024);
  _Float16*  Hb       = (_Float16*)(w + 4227328);
  float*     Ob       = (float*)(w + 21004544);

  k_route<<<1, 256, 0, stream>>>(indices, offs, rowtok, pairslot);
  k_gather<<<NP, 256, 0, stream>>>(x, rowtok, Xg);

  dim3 g1(NP / BM, NH / BN, NE);   // (32, 16, 8); inactive tail tiles exit
  k_moe_gemm<ND, NH, true><<<g1, 256, 0, stream>>>(Xg, keys, key_bias, offs, (void*)Hb);

  dim3 g2(NP / BM, ND / BN, NE);   // (32, 4, 8)
  k_moe_gemm<NH, ND, false><<<g2, 256, 0, stream>>>(Hb, values, value_bias, offs, (void*)Ob);

  k_combine<<<NT, 256, 0, stream>>>(Ob, pairslot, weights, out);
}
